// BiGRUCRF_86062554677931
// MI455X (gfx1250) — matrix operand order repaired and checked by reading, not compiled, not run on hardware
//
#include <hip/hip_runtime.h>
#include <hip/hip_bf16.h>

#define Bn 256
#define Tn 512
#define Dn 256
#define Hn 64
#define H3 192
#define Kn 32
#define MR (Bn * Tn)   // 131072 flattened rows

typedef __attribute__((ext_vector_type(16))) __bf16 v16bf;
typedef __attribute__((ext_vector_type(8)))  float  v8f;

union V16BF { v16bf v; unsigned short u[16]; uint4 q[2]; };

#if defined(__AMDGCN__) && __has_builtin(__builtin_amdgcn_global_load_async_to_lds_b128) && __has_builtin(__builtin_amdgcn_s_wait_asynccnt)
#define HAVE_ASYNC_LDS 1
typedef int a_v4i __attribute__((__vector_size__(16)));
typedef __attribute__((address_space(1))) a_v4i* gbl_v4i;   // 64-bit global ptr
typedef __attribute__((address_space(3))) a_v4i* lds_v4i;   // 32-bit LDS ptr
#else
#define HAVE_ASYNC_LDS 0
#endif

__device__ inline unsigned short f2bf(float f) {
  union { float f; unsigned int u; } x; x.f = f;
  unsigned int r = x.u + 0x7FFFu + ((x.u >> 16) & 1u);   // round-to-nearest-even
  return (unsigned short)(r >> 16);
}

// A fragment (16x32 bf16) from a row-major [16 x lda] bf16 buffer, K offset k0.
__device__ inline v16bf frag_a(const unsigned short* s, int lda, int k0, int lane) {
  int row = lane & 15;
  int kb  = k0 + ((lane >> 4) << 3);
  V16BF r;
  r.q[0] = *(const uint4*)(s + row * lda + kb);
  r.q[1] = *(const uint4*)(s + row * lda + kb + 16);
  return r.v;
}

// B fragment (32x16 bf16) from W^T stored row-major [N][ldk] at tile (n0, k0).
__device__ inline v16bf frag_b(const unsigned short* wt, int ldk, int n0, int k0, int lane) {
  int n  = n0 + (lane & 15);
  int kb = k0 + ((lane >> 4) << 3);
  V16BF r;
  const uint4* p = (const uint4*)(wt + n * ldk + kb);
  r.q[0] = p[0];
  r.q[1] = p[2];
  return r.v;
}

__device__ inline v8f wmma_bf16(v16bf a, v16bf b, v8f c) {
  return __builtin_amdgcn_wmma_f32_16x16x32_bf16(false, a, false, b, (short)0, c, false, false);
}

// ---------------- weight prep: transpose + f32 -> bf16 ----------------
__global__ void prep_weights_kernel(const float* kfw, const float* kbw,
                                    const float* rfw, const float* rbw,
                                    const float* dk,
                                    unsigned short* WtFw, unsigned short* WtBw,
                                    unsigned short* RtFw, unsigned short* RtBw,
                                    unsigned short* Dt) {
  int stride = gridDim.x * blockDim.x;
  int tid = blockIdx.x * blockDim.x + threadIdx.x;
  for (int i = tid; i < H3 * Dn; i += stride) {      // Wt[n][d] = K[d][n]
    int n = i / Dn, d = i % Dn;
    WtFw[i] = f2bf(kfw[d * H3 + n]);
    WtBw[i] = f2bf(kbw[d * H3 + n]);
  }
  for (int i = tid; i < H3 * Hn; i += stride) {      // Rt[n][h] = R[h][n]
    int n = i / Hn, h = i % Hn;
    RtFw[i] = f2bf(rfw[h * H3 + n]);
    RtBw[i] = f2bf(rbw[h * H3 + n]);
  }
  for (int i = tid; i < Kn * 2 * Hn; i += stride) {  // Dt[k][h] = D[h][k]
    int k = i / (2 * Hn), h = i % (2 * Hn);
    Dt[i] = f2bf(dk[h * Kn + k]);
  }
}

// ---------------- input projection: Xp = X @ W + b0 ----------------
// One block stages a 16x256 X tile ONCE and produces all 6 (dir, n-block)
// outputs from it: A fragment is reused across 6 WMMAs per K-step, so X HBM
// traffic is 134 MB total instead of 6x that.
__global__ void __launch_bounds__(128) xproj_kernel(const float* X,
                                                    const unsigned short* WtFw,
                                                    const unsigned short* WtBw,
                                                    const float* bFw, const float* bBw,
                                                    float* XpFw, float* XpBw) {
  __shared__ unsigned short sh[16 * Dn];  // 16 rows of X in bf16
  int tid = threadIdx.x;
  int m0 = blockIdx.x * 16;
  for (int i = tid; i < 16 * Dn; i += 128) {
    int r = i >> 8;
    int d = i & (Dn - 1);
    sh[i] = f2bf(X[(size_t)(m0 + r) * Dn + d]);
  }
  __syncthreads();
  int lane = tid & 31;
  int wave = tid >> 5;

  v8f acc[6] = {};                       // j = dir*3 + nb
#pragma unroll
  for (int kk = 0; kk < Dn / 32; ++kk) {
    v16bf a = frag_a(sh, Dn, kk * 32, lane);
#pragma unroll
    for (int j = 0; j < 6; ++j) {
      const unsigned short* Wt = (j < 3) ? WtFw : WtBw;
      int n0 = (j % 3) * 64 + wave * 16;
      v16bf b = frag_b(Wt, Dn, n0, kk * 32, lane);
      acc[j] = wmma_bf16(a, b, acc[j]);
    }
  }
  int rbase = m0 + ((lane >> 4) << 3);
#pragma unroll
  for (int j = 0; j < 6; ++j) {
    int dir = j / 3;
    int col = (j % 3) * 64 + wave * 16 + (lane & 15);
    const float* bias = dir ? bBw : bFw;
    float* Xp         = dir ? XpBw : XpFw;
    float bv = bias[col];                // bias row 0 (input bias)
#pragma unroll
    for (int v = 0; v < 8; ++v)
      Xp[(size_t)(rbase + v) * H3 + col] = acc[j][v] + bv;
  }
}

// ---------------- persistent GRU scan, both dirs concurrently ----------------
// Wave w owns h-columns [w*16, w*16+16) of ALL three gates, so z/r/c accumulators
// and h_prev stay in the WMMA C-register layout; only the bf16 h tile goes via LDS.
__global__ void __launch_bounds__(128) gru_scan_kernel(
    const float* XpFw, const float* XpBw, const int* mask,
    const unsigned short* RtFw, const unsigned short* RtBw,
    const float* bFw, const float* bBw,
    unsigned short* HbfFw, unsigned short* HbfBw) {
  int dir = blockIdx.y;
  const float* Xp          = dir ? XpBw : XpFw;
  const unsigned short* Rt = dir ? RtBw : RtFw;
  const float* bias        = dir ? bBw : bFw;
  unsigned short* Hout     = dir ? HbfBw : HbfFw;

  __shared__ unsigned short h_bf[16 * Hn];   // 2 KB bf16 state tile

  int tid = threadIdx.x, lane = tid & 31, wave = tid >> 5;
  int b0 = blockIdx.x * 16;

  for (int i = tid; i < 16 * Hn; i += 128) h_bf[i] = 0;

  // Recurrent weights in registers: gate g, K-half kh -> cols g*64 + wave*16.
  v16bf Bf[3][2];
#pragma unroll
  for (int g = 0; g < 3; ++g) {
    Bf[g][0] = frag_b(Rt, Hn, g * 64 + wave * 16, 0, lane);
    Bf[g][1] = frag_b(Rt, Hn, g * 64 + wave * 16, 32, lane);
  }
  int hcol = wave * 16 + (lane & 15);
  float brz = bias[H3 + hcol];              // recurrent bias (row 1)
  float brr = bias[H3 + Hn + hcol];
  float brc = bias[H3 + 2 * Hn + hcol];

  float hprev[8];
#pragma unroll
  for (int v = 0; v < 8; ++v) hprev[v] = 0.f;

  int rbase = (lane >> 4) << 3;             // C-layout row base: 0 or 8
  __syncthreads();

  for (int step = 0; step < Tn; ++step) {
    int t  = dir ? (Tn - 1 - step) : step;
    int tn = dir ? (t - 1) : (t + 1);       // next step's t (for prefetch)

    // Gate inputs: independent of h -> issue first; prefetch next step's rows.
    float xz[8], xr[8], xh[8];
#pragma unroll
    for (int v = 0; v < 8; ++v) {
      const float* xp = Xp + ((size_t)(b0 + rbase + v) * Tn + t) * H3;
      xz[v] = xp[hcol];
      xr[v] = xp[Hn + hcol];
      xh[v] = xp[2 * Hn + hcol];
      if (tn >= 0 && tn < Tn)
        __builtin_prefetch(Xp + ((size_t)(b0 + rbase + v) * Tn + tn) * H3 + hcol, 0, 1);
    }

    // hp = h @ rec : 2 WMMAs per gate (K = 64)
    v16bf a0 = frag_a(h_bf, Hn, 0, lane);
    v16bf a1 = frag_a(h_bf, Hn, 32, lane);
    v8f az = {}, ar = {}, ac = {};
    az = wmma_bf16(a0, Bf[0][0], az); az = wmma_bf16(a1, Bf[0][1], az);
    ar = wmma_bf16(a0, Bf[1][0], ar); ar = wmma_bf16(a1, Bf[1][1], ar);
    ac = wmma_bf16(a0, Bf[2][0], ac); ac = wmma_bf16(a1, Bf[2][1], ac);
    __syncthreads();                         // all waves done reading h_bf (WAR)

#pragma unroll
    for (int v = 0; v < 8; ++v) {
      int row = rbase + v;
      int b = b0 + row;
      float z = 1.f / (1.f + expf(-(xz[v] + az[v] + brz)));
      float r = 1.f / (1.f + expf(-(xr[v] + ar[v] + brr)));
      float c = tanhf(xh[v] + r * (ac[v] + brc));
      float hn = z * hprev[v] + (1.f - z) * c;
      if (mask[b * Tn + t] <= 0) hn = hprev[v];
      hprev[v] = hn;
      unsigned short hb = f2bf(hn);
      h_bf[row * Hn + hcol] = hb;            // conflict-free: unique (row,col) owner
      Hout[((size_t)b * Tn + t) * Hn + hcol] = hb;
    }
    __syncthreads();                         // h_bf ready for next step (RAW)
  }
}

// ---------------- dense + boundary epilogue -> potentials (bf16 hidden in) ----------------
__global__ void __launch_bounds__(128) dense_kernel(const unsigned short* Hfw,
                                                    const unsigned short* Hbw,
                                                    const unsigned short* Dt,
                                                    const float* dbias, const float* lb,
                                                    const float* rbnd, float* pot) {
  __shared__ unsigned short sh[32 * 128];  // 32 rows of concat(fw,bw), bf16
  int tid = threadIdx.x;
  int m0 = blockIdx.x * 32;
  // Stage: pure byte copy global -> LDS (no conversion). Use the gfx1250
  // async-to-LDS path (ASYNCcnt) when the toolchain exposes it.
  for (int i = tid; i < 32 * 16; i += 128) {
    int r = i >> 4;
    int c = (i & 15) * 8;
    const unsigned short* src = (c < Hn) ? (Hfw + (size_t)(m0 + r) * Hn + c)
                                         : (Hbw + (size_t)(m0 + r) * Hn + (c - Hn));
#if HAVE_ASYNC_LDS
    __builtin_amdgcn_global_load_async_to_lds_b128(
        (gbl_v4i)(unsigned long long)src,
        (lds_v4i)(unsigned)(unsigned long long)(sh + r * 128 + c),
        0, 0);
#else
    *(uint4*)(sh + r * 128 + c) = *(const uint4*)src;
#endif
  }
#if HAVE_ASYNC_LDS
  __builtin_amdgcn_s_wait_asynccnt(0);
#endif
  __syncthreads();
  int lane = tid & 31;
  int wave = tid >> 5;
  int mt = wave >> 1, nt = wave & 1;
  v8f acc = {};
#pragma unroll
  for (int kk = 0; kk < 4; ++kk) {
    v16bf a = frag_a(sh + mt * 16 * 128, 128, kk * 32, lane);
    v16bf b = frag_b(Dt, 128, nt * 16, kk * 32, lane);
    acc = wmma_bf16(a, b, acc);
  }
  int col = nt * 16 + (lane & 15);
  float bv = dbias[col];
  int rbase = m0 + mt * 16 + ((lane >> 4) << 3);
#pragma unroll
  for (int v = 0; v < 8; ++v) {
    int row = rbase + v;
    int t = row & (Tn - 1);
    float val = acc[v] + bv;
    if (t == 0) val += lb[col];
    if (t == Tn - 1) val += rbnd[col];
    pot[(size_t)row * Kn + col] = val;
  }
}

// ---------------- Viterbi: lane == state (K=32 == wave32) ----------------
__global__ void __launch_bounds__(256) viterbi_kernel(const float* pot, const float* chain,
                                                      const int* mask, unsigned char* bp,
                                                      float* dec, float* slen,
                                                      float* chain_out) {
  __shared__ float sc[8][32];
  int tid = threadIdx.x;
  int lane = tid & 31;
  int wave = tid >> 5;
  int b = blockIdx.x * 8 + wave;

  if (blockIdx.x == 0)
    for (int i = tid; i < Kn * Kn; i += 256) chain_out[i] = chain[i];

  float tc[32];  // column `lane` of trans, in registers
#pragma unroll
  for (int kp = 0; kp < 32; ++kp) tc[kp] = chain[kp * Kn + lane];

  const float* pb = pot + (size_t)b * Tn * Kn;
  unsigned char* bpb = bp + (size_t)b * Kn;
  float score = pb[lane];
  for (int t = 1; t < Tn; ++t) {
    float best = -3.4e38f;
    int arg = 0;
#pragma unroll
    for (int kp = 0; kp < 32; ++kp) {
      float s = __shfl(score, kp, 32);
      float v = s + tc[kp];
      if (v > best) { best = v; arg = kp; }   // first-max tie-break, matches argmax
    }
    score = best + pb[t * Kn + lane];
    bpb[(size_t)(t - 1) * (Bn * Kn) + lane] = (unsigned char)arg;
  }
  sc[wave][lane] = score;
  __syncthreads();

  // sequence lengths (wave reduction over mask)
  int s = 0;
  for (int t = lane; t < Tn; t += 32) s += (mask[b * Tn + t] > 0) ? 1 : 0;
#pragma unroll
  for (int off = 16; off > 0; off >>= 1) s += __shfl_down(s, off, 32);

  if (lane == 0) {
    slen[b] = (float)s;
    float best = sc[wave][0]; int tag = 0;
    for (int k = 1; k < 32; ++k)
      if (sc[wave][k] > best) { best = sc[wave][k]; tag = k; }
    dec[(size_t)b * Tn + (Tn - 1)] = (float)tag;
    for (int t = Tn - 2; t >= 0; --t) {
      tag = bpb[(size_t)t * (Bn * Kn) + tag];
      dec[(size_t)b * Tn + t] = (float)tag;
    }
  }
}

extern "C" void kernel_launch(void* const* d_in, const int* in_sizes, int n_in,
                              void* d_out, int out_size, void* d_ws, size_t ws_size,
                              hipStream_t stream) {
  (void)in_sizes; (void)n_in; (void)out_size; (void)ws_size;
  const float* X     = (const float*)d_in[0];
  const int*   mask  = (const int*)d_in[1];
  const float* kfw   = (const float*)d_in[2];
  const float* rfw   = (const float*)d_in[3];
  const float* bfw   = (const float*)d_in[4];
  const float* kbw   = (const float*)d_in[5];
  const float* rbw   = (const float*)d_in[6];
  const float* bbw   = (const float*)d_in[7];
  const float* dk    = (const float*)d_in[8];
  const float* db    = (const float*)d_in[9];
  const float* chain = (const float*)d_in[10];
  const float* lb    = (const float*)d_in[11];
  const float* rbnd  = (const float*)d_in[12];

  char* ws = (char*)d_ws;
  size_t off = 0;
  auto alloc = [&](size_t bytes) {
    void* p = ws + off;
    off = (off + bytes + 255) & ~(size_t)255;
    return p;
  };
  unsigned short* WtFw = (unsigned short*)alloc((size_t)H3 * Dn * 2);
  unsigned short* WtBw = (unsigned short*)alloc((size_t)H3 * Dn * 2);
  unsigned short* RtFw = (unsigned short*)alloc((size_t)H3 * Hn * 2);
  unsigned short* RtBw = (unsigned short*)alloc((size_t)H3 * Hn * 2);
  unsigned short* Dt   = (unsigned short*)alloc((size_t)Kn * 2 * Hn * 2);
  float* XpFw = (float*)alloc((size_t)MR * H3 * 4);
  float* XpBw = (float*)alloc((size_t)MR * H3 * 4);
  unsigned short* HbfFw = (unsigned short*)alloc((size_t)MR * Hn * 2);
  unsigned short* HbfBw = (unsigned short*)alloc((size_t)MR * Hn * 2);
  unsigned char* bp = (unsigned char*)alloc((size_t)(Tn - 1) * Bn * Kn);

  float* out = (float*)d_out;
  float* dec  = out;
  float* pot  = out + (size_t)Bn * Tn;
  float* slen = pot + (size_t)Bn * Tn * Kn;
  float* chn  = slen + Bn;

  prep_weights_kernel<<<64, 256, 0, stream>>>(kfw, kbw, rfw, rbw, dk,
                                              WtFw, WtBw, RtFw, RtBw, Dt);
  xproj_kernel<<<MR / 16, 128, 0, stream>>>(X, WtFw, WtBw, bfw, bbw, XpFw, XpBw);
  dim3 gg(Bn / 16, 2);
  gru_scan_kernel<<<gg, 128, 0, stream>>>(XpFw, XpBw, mask, RtFw, RtBw,
                                          bfw, bbw, HbfFw, HbfBw);
  dense_kernel<<<MR / 32, 128, 0, stream>>>(HbfFw, HbfBw, Dt, db, lb, rbnd, pot);
  viterbi_kernel<<<Bn / 8, 256, 0, stream>>>(pot, chain, mask, bp, dec, slen, chn);
}
